// PerformerBiLSTM_61632780697956
// MI455X (gfx1250) — matrix operand order repaired and checked by reading, not compiled, not run on hardware
//
#include <hip/hip_runtime.h>
#include <hip/hip_bf16.h>
#include <math.h>

typedef __attribute__((ext_vector_type(16))) _Float16 v16h;
typedef __attribute__((ext_vector_type(8)))  _Float16 v8h;
typedef __attribute__((ext_vector_type(8)))  float    v8f;

#define B_  4
#define S_  4096
#define D_  512
#define H_  8
#define KD_ 64
#define R_  256
#define EPS_   1e-3f
#define SCALE_ 0.125f   // 1/sqrt(64)

// ---------------------------------------------------------------------------
// WMMA helpers (CDNA5 ISA 7.12.2, wave32). All operands are arranged so that
// each lane's 16 halves are two aligned 16-byte chunks -> global_load_b128.
// ---------------------------------------------------------------------------

__device__ __forceinline__ v8f wmma_f16(v16h a, v16h b, v8f c) {
  return __builtin_amdgcn_wmma_f32_16x16x32_f16(
      false, a, false, b, (short)0, c, false, false);
}

__device__ __forceinline__ v16h concat16(v8h lo, v8h hi) {
  return __builtin_shufflevector(lo, hi, 0, 1, 2, 3, 4, 5, 6, 7,
                                 8, 9, 10, 11, 12, 13, 14, 15);
}

// A-matrix 16x32 f16 tile from row-major source (lda in elements, multiple of 8).
// lanes 0-15: K = k0+{0..7, 16..23}; lanes 16-31: +8.  Two b128 loads per lane.
__device__ __forceinline__ v16h load_A16x32(const _Float16* __restrict__ A,
                                            size_t lda, int k0, int lane) {
  const _Float16* p = A + (size_t)(lane & 15) * lda + k0 + ((lane < 16) ? 0 : 8);
  return concat16(*(const v8h*)p, *(const v8h*)(p + 16));
}

// B-matrix 32x16 tile from K-contiguous storage BT[n*ld + k] ("transposed B").
// lanes 0-15: K = k0+{0..7, 16..23}; lanes 16-31: +8; n = n0 + lane&15.
__device__ __forceinline__ v16h load_B32x16T(const _Float16* __restrict__ BT,
                                             size_t ld, int k0, int n0, int lane) {
  const _Float16* p = BT + (size_t)(n0 + (lane & 15)) * ld + k0 +
                      ((lane < 16) ? 0 : 8);
  return concat16(*(const v8h*)p, *(const v8h*)(p + 16));
}

// ---------------------------------------------------------------------------
// Elementwise utility kernels
// ---------------------------------------------------------------------------

__global__ void cvt_f32_to_f16_kernel(const float* __restrict__ src,
                                      _Float16* __restrict__ dst, int n) {
  int i = blockIdx.x * blockDim.x + threadIdx.x;
  if (i < n) dst[i] = (_Float16)src[i];
}

// W [D,D] row-major f32 -> WT [n][k] f16 (K-contiguous for WMMA B operand)
__global__ void cvt_transpose_kernel(const float* __restrict__ src,
                                     _Float16* __restrict__ dst) {
  int i = blockIdx.x * blockDim.x + threadIdx.x;
  if (i >= D_ * D_) return;
  int n = i % D_, k = i / D_;
  dst[(size_t)n * D_ + k] = (_Float16)src[i];
}

// kv f32 [BH,R,KD] -> kvT f16 [BH,KD,R]
__global__ void cvt_kvT_kernel(const float* __restrict__ kvf,
                               _Float16* __restrict__ kvT) {
  int i = blockIdx.x * blockDim.x + threadIdx.x;
  if (i >= B_ * H_ * R_ * KD_) return;
  int d = i % KD_;
  int t = i / KD_;
  int r = t % R_;
  int bh = t / R_;
  kvT[((size_t)bh * KD_ + d) * R_ + r] = (_Float16)kvf[i];
}

__global__ void zero_f32_kernel(float* __restrict__ p, int n) {
  int i = blockIdx.x * blockDim.x + threadIdx.x;
  if (i < n) p[i] = 0.0f;
}

// ---------------------------------------------------------------------------
// Generic f16 GEMM: C[M,N] = A[M,K] @ BT[N,K]^T.  block = 8 waves,
// block tile 16(M) x 128(N).  The shared 16x32 A tile is staged into LDS once
// per K-step with GLOBAL_LOAD_ASYNC_TO_LDS_B128 (ASYNCcnt) and read back by
// all 8 waves as ds_load_b128 pairs.
// OUT_MODE: 0 = f32 + bias, 1 = f16 row-major, 2 = f16 in [B,H,KD,S] layout
// ---------------------------------------------------------------------------

template <int OUT_MODE>
__global__ __launch_bounds__(256) void gemm_f16_kernel(
    const _Float16* __restrict__ A, const _Float16* __restrict__ BT,
    const float* __restrict__ bias, void* __restrict__ Cout,
    int M, int N, int K) {
  __shared__ _Float16 As[16][32];  // sole LDS object -> offset 0

  int lane = threadIdx.x & 31;
  int wave = threadIdx.x >> 5;
  int n0 = blockIdx.x * 128 + wave * 16;
  int m0 = blockIdx.y * 16;
  const _Float16* Ab = A + (size_t)m0 * K;

  v8f acc;
#pragma unroll
  for (int r = 0; r < 8; ++r) acc[r] = 0.0f;

  for (int k0 = 0; k0 < K; k0 += 32) {
    // async-stage the 16x32 A tile (1 KB): waves 0-1, one b128 per lane
    if (threadIdx.x < 64) {
      int r = threadIdx.x >> 2;        // row 0..15
      int c = (threadIdx.x & 3) * 8;   // col 0,8,16,24
      unsigned lds_off = (unsigned)((r * 32 + c) * sizeof(_Float16));
      unsigned long long ga =
          (unsigned long long)(Ab + (size_t)r * K + k0 + c);
      asm volatile("global_load_async_to_lds_b128 %0, %1, off"
                   :: "v"(lds_off), "v"(ga) : "memory");
    }
    asm volatile("s_wait_asynccnt 0" ::: "memory");
    __syncthreads();

    const _Float16* p = &As[lane & 15][(lane < 16) ? 0 : 8];
    v16h a = concat16(*(const v8h*)p, *(const v8h*)(p + 16));  // 2x ds_load_b128
    v16h b = load_B32x16T(BT, (size_t)K, k0, n0, lane);
    acc = wmma_f16(a, b, acc);
    __syncthreads();  // protect LDS tile before next-iteration overwrite
  }

  int n = n0 + (lane & 15);
  int mb = m0 + ((lane < 16) ? 0 : 8);
  if (OUT_MODE == 0) {
    float bv = bias[n];
    float* C = (float*)Cout;
#pragma unroll
    for (int r = 0; r < 8; ++r) C[(size_t)(mb + r) * N + n] = acc[r] + bv;
  } else if (OUT_MODE == 1) {
    _Float16* C = (_Float16*)Cout;
#pragma unroll
    for (int r = 0; r < 8; ++r) C[(size_t)(mb + r) * N + n] = (_Float16)acc[r];
  } else {
    // vT layout [B,H,KD,S]: col n -> (h,d); rows mb..mb+7 -> consecutive s
    _Float16* C = (_Float16*)Cout;
    int h = n >> 6, d = n & 63;
    int b = mb / S_, s = mb % S_;
    _Float16* q = C + ((size_t)(b * H_ + h) * KD_ + d) * S_ + s;
#pragma unroll
    for (int r = 0; r < 8; ++r) q[r] = (_Float16)acc[r];  // contiguous in s
  }
}

// ---------------------------------------------------------------------------
// Random-feature kernel: proj = (t @ rf^T) * SCALE; f = exp(proj - rowmax)+eps
// One block = (b, 16-row s-tile, h); 8 waves x 2 N-tiles cover R=256.
// TRANS=false -> out[b,s,h,r]; TRANS=true -> out[b,h,r,s]
// ---------------------------------------------------------------------------

template <bool TRANS>
__global__ __launch_bounds__(256) void feature_kernel(
    const _Float16* __restrict__ T16,   // [B,S,H,KD]
    const _Float16* __restrict__ rf16,  // [R,KD] == BT[n][k]
    _Float16* __restrict__ out) {
  __shared__ float tile[16][R_ + 8];
  __shared__ float rowmax[16];

  int lane = threadIdx.x & 31;
  int wave = threadIdx.x >> 5;
  int bx = blockIdx.x;
  int h = bx % H_;
  int t = bx / H_;
  int sblk = t % (S_ / 16);
  int b = t / (S_ / 16);
  int s0 = sblk * 16;

  const _Float16* Ab = T16 + ((size_t)(b * S_ + s0) * H_ + h) * KD_;
  int n0 = wave * 32;

  v8f acc0, acc1;
#pragma unroll
  for (int r = 0; r < 8; ++r) { acc0[r] = 0.0f; acc1[r] = 0.0f; }

#pragma unroll
  for (int k0 = 0; k0 < KD_; k0 += 32) {
    v16h a  = load_A16x32(Ab, (size_t)H_ * KD_, k0, lane);
    v16h b0 = load_B32x16T(rf16, (size_t)KD_, k0, n0, lane);
    v16h b1 = load_B32x16T(rf16, (size_t)KD_, k0, n0 + 16, lane);
    acc0 = wmma_f16(a, b0, acc0);
    acc1 = wmma_f16(a, b1, acc1);
  }

  int n  = lane & 15;
  int mb = (lane < 16) ? 0 : 8;
#pragma unroll
  for (int r = 0; r < 8; ++r) {
    tile[mb + r][n0 + n]      = acc0[r] * SCALE_;
    tile[mb + r][n0 + 16 + n] = acc1[r] * SCALE_;
  }
  __syncthreads();

  if (threadIdx.x < 16) {
    float m = -3.4e38f;
    for (int c = 0; c < R_; ++c) m = fmaxf(m, tile[threadIdx.x][c]);
    rowmax[threadIdx.x] = m;
  }
  __syncthreads();

  int row = threadIdx.x >> 4;
  int cb  = threadIdx.x & 15;
  float mx = rowmax[row];
#pragma unroll
  for (int i = 0; i < 16; ++i) {
    int c = cb + 16 * i;
    float v = expf(tile[row][c] - mx) + EPS_;
    if (TRANS)
      out[((size_t)(b * H_ + h) * R_ + c) * S_ + (s0 + row)] = (_Float16)v;
    else
      out[((size_t)(b * S_ + s0 + row) * H_ + h) * R_ + c] = (_Float16)v;
  }
}

// ---------------------------------------------------------------------------
// kv = kf^T @ v per (b,h): C[R=256, KD=64], S=4096 reduction split into 8
// chunks (256 blocks), merged with global_atomic_add_f32.
// A = kfT [B,H,R,S] (K-contiguous); B = vT [B,H,KD,S] (K-contiguous).
// ---------------------------------------------------------------------------

__global__ __launch_bounds__(256) void kv_kernel(
    const _Float16* __restrict__ kfT,  // [B,H,R,S]
    const _Float16* __restrict__ vT,   // [B,H,KD,S]
    float* __restrict__ kv) {          // [B,H,R,KD]
  const int NCH = 8, SCH = S_ / NCH;
  int lane = threadIdx.x & 31;
  int wave = threadIdx.x >> 5;
  int bx = blockIdx.x;
  int ch = bx % NCH;
  int h  = (bx / NCH) % H_;
  int b  = bx / (NCH * H_);

  const _Float16* Ab = kfT + (size_t)(b * H_ + h) * R_ * S_;
  const _Float16* Bb = vT + (size_t)(b * H_ + h) * KD_ * S_;

  v8f acc[2][4];
#pragma unroll
  for (int mt = 0; mt < 2; ++mt)
#pragma unroll
    for (int nt = 0; nt < 4; ++nt)
#pragma unroll
      for (int r = 0; r < 8; ++r) acc[mt][nt][r] = 0.0f;

  for (int s0 = ch * SCH; s0 < (ch + 1) * SCH; s0 += 32) {
    v16h bm[4];
#pragma unroll
    for (int nt = 0; nt < 4; ++nt)
      bm[nt] = load_B32x16T(Bb, (size_t)S_, s0, nt * 16, lane);
#pragma unroll
    for (int mt = 0; mt < 2; ++mt) {
      int m0 = (wave * 2 + mt) * 16;
      v16h a = load_A16x32(Ab + (size_t)m0 * S_, (size_t)S_, s0, lane);
#pragma unroll
      for (int nt = 0; nt < 4; ++nt)
        acc[mt][nt] = wmma_f16(a, bm[nt], acc[mt][nt]);
    }
  }

  float* Cb = kv + (size_t)(b * H_ + h) * R_ * KD_;
  int n  = lane & 15;
  int mo = (lane < 16) ? 0 : 8;
#pragma unroll
  for (int mt = 0; mt < 2; ++mt)
#pragma unroll
    for (int nt = 0; nt < 4; ++nt)
#pragma unroll
      for (int r = 0; r < 8; ++r)
        atomicAdd(&Cb[(size_t)((wave * 2 + mt) * 16 + mo + r) * KD_ + nt * 16 + n],
                  acc[mt][nt][r]);
}

// ksum[b,h,r] = sum_s kfT[b,h,r,s]  (vectorized f16 reads, f32 accumulate)
__global__ void ksum_kernel(const _Float16* __restrict__ kfT,
                            float* __restrict__ ksum) {
  int i = blockIdx.x * blockDim.x + threadIdx.x;
  if (i >= B_ * H_ * R_) return;
  const _Float16* p = kfT + (size_t)i * S_;
  float s = 0.0f;
  for (int j = 0; j < S_; j += 8) {
    v8h t = *(const v8h*)(p + j);
#pragma unroll
    for (int e = 0; e < 8; ++e) s += (float)t[e];
  }
  ksum[i] = s;
}

// ---------------------------------------------------------------------------
// num/den kernel: num = qf @ kv; den = qf.ksum + eps; attn = num/den -> f16
// block = 4 waves: 16 rows(s) x 64 cols(d); K loop over R=256.
// B = kvT [B,H,KD,R] (K-contiguous).
// ---------------------------------------------------------------------------

__global__ __launch_bounds__(128) void num_kernel(
    const _Float16* __restrict__ qf,   // [B,S,H,R]
    const _Float16* __restrict__ kvT,  // [B,H,KD,R]
    const float* __restrict__ ksum,    // [B,H,R]
    _Float16* __restrict__ attn) {     // [B,S,D]
  __shared__ float den[16];
  int lane = threadIdx.x & 31;
  int wave = threadIdx.x >> 5;
  int bx = blockIdx.x;
  int h = bx % H_;
  int t = bx / H_;
  int sblk = t % (S_ / 16);
  int b = t / (S_ / 16);
  int s0 = sblk * 16;

  const _Float16* Ab = qf + ((size_t)(b * S_ + s0) * H_ + h) * R_;
  const _Float16* Bb = kvT + (size_t)(b * H_ + h) * KD_ * R_;

  if (threadIdx.x < 16) {
    const _Float16* q = Ab + (size_t)threadIdx.x * (H_ * R_);
    const float* ks = ksum + (size_t)(b * H_ + h) * R_;
    float d = EPS_;
    for (int r = 0; r < R_; ++r) d += (float)q[r] * ks[r];
    den[threadIdx.x] = d;
  }

  v8f acc;
#pragma unroll
  for (int r = 0; r < 8; ++r) acc[r] = 0.0f;
  int n0 = wave * 16;
#pragma unroll
  for (int k0 = 0; k0 < R_; k0 += 32) {
    v16h a  = load_A16x32(Ab, (size_t)H_ * R_, k0, lane);
    v16h bm = load_B32x16T(Bb, (size_t)R_, k0, n0, lane);
    acc = wmma_f16(a, bm, acc);
  }
  __syncthreads();

  int n  = n0 + (lane & 15);
  int mo = (lane < 16) ? 0 : 8;
#pragma unroll
  for (int r = 0; r < 8; ++r) {
    float v = acc[r] / den[mo + r];
    attn[(size_t)(b * S_ + s0 + mo + r) * D_ + h * KD_ + n] = (_Float16)v;
  }
}

// ---------------------------------------------------------------------------
// Host-side launcher
// ---------------------------------------------------------------------------

extern "C" void kernel_launch(void* const* d_in, const int* in_sizes, int n_in,
                              void* d_out, int out_size, void* d_ws, size_t ws_size,
                              hipStream_t stream) {
  const float* x  = (const float*)d_in[0];
  const float* Wq = (const float*)d_in[1];
  const float* Wk = (const float*)d_in[2];
  const float* Wv = (const float*)d_in[3];
  const float* rf = (const float*)d_in[4];
  const float* Wo = (const float*)d_in[5];
  const float* bo = (const float*)d_in[6];

  const int NX  = B_ * S_ * D_;        // 8388608
  const int NW  = D_ * D_;             // 262144
  const int NRF = R_ * KD_;            // 16384
  const int NQF = B_ * S_ * H_ * R_;   // 33554432
  const int NKV = B_ * H_ * R_ * KD_;  // 524288
  const int NKS = B_ * H_ * R_;        // 8192

  char* ws = (char*)d_ws;
  size_t off = 0;
  auto alloc = [&](size_t bytes) -> void* {
    void* p = ws + off;
    off += (bytes + 255) & ~(size_t)255;
    return p;
  };

  _Float16* x16   = (_Float16*)alloc((size_t)NX * 2);
  _Float16* WqT   = (_Float16*)alloc((size_t)NW * 2);
  _Float16* WkT   = (_Float16*)alloc((size_t)NW * 2);
  _Float16* WvT   = (_Float16*)alloc((size_t)NW * 2);
  _Float16* WoT   = (_Float16*)alloc((size_t)NW * 2);
  _Float16* rf16  = (_Float16*)alloc((size_t)NRF * 2);
  _Float16* q16   = (_Float16*)alloc((size_t)NX * 2);
  _Float16* k16   = (_Float16*)alloc((size_t)NX * 2);
  _Float16* vT    = (_Float16*)alloc((size_t)NX * 2);
  _Float16* qf16  = (_Float16*)alloc((size_t)NQF * 2);
  _Float16* kfT16 = (_Float16*)alloc((size_t)NQF * 2);
  float*    kvf   = (float*)alloc((size_t)NKV * 4);
  _Float16* kvT16 = (_Float16*)alloc((size_t)NKV * 2);
  float*    ksum  = (float*)alloc((size_t)NKS * 4);
  _Float16* attn16 = x16;  // x16 dead after QKV projections; reuse

  // 1) conversions (weights transposed to K-contiguous for WMMA B operands)
  cvt_f32_to_f16_kernel<<<(NX + 255) / 256, 256, 0, stream>>>(x, x16, NX);
  cvt_transpose_kernel<<<(NW + 255) / 256, 256, 0, stream>>>(Wq, WqT);
  cvt_transpose_kernel<<<(NW + 255) / 256, 256, 0, stream>>>(Wk, WkT);
  cvt_transpose_kernel<<<(NW + 255) / 256, 256, 0, stream>>>(Wv, WvT);
  cvt_transpose_kernel<<<(NW + 255) / 256, 256, 0, stream>>>(Wo, WoT);
  cvt_f32_to_f16_kernel<<<(NRF + 255) / 256, 256, 0, stream>>>(rf, rf16, NRF);

  // 2) QKV projections (16384x512 @ 512x512); v stored transposed [B,H,KD,S]
  dim3 ggrid(D_ / 128, (B_ * S_) / 16);
  gemm_f16_kernel<1><<<ggrid, 256, 0, stream>>>(x16, WqT, nullptr, q16, B_ * S_, D_, D_);
  gemm_f16_kernel<1><<<ggrid, 256, 0, stream>>>(x16, WkT, nullptr, k16, B_ * S_, D_, D_);
  gemm_f16_kernel<2><<<ggrid, 256, 0, stream>>>(x16, WvT, nullptr, vT, B_ * S_, D_, D_);

  // 3) random features (qf row-major, kf transposed for the kv GEMM)
  int fblocks = B_ * (S_ / 16) * H_;  // 8192
  feature_kernel<false><<<fblocks, 256, 0, stream>>>(q16, rf16, qf16);
  feature_kernel<true><<<fblocks, 256, 0, stream>>>(k16, rf16, kfT16);

  // 4) kv = kf^T @ v (chunked over S, atomic merge) + ksum; kv -> [B,H,KD,R] f16
  zero_f32_kernel<<<(NKV + 255) / 256, 256, 0, stream>>>(kvf, NKV);
  kv_kernel<<<B_ * H_ * 8, 256, 0, stream>>>(kfT16, vT, kvf);
  ksum_kernel<<<(NKS + 255) / 256, 256, 0, stream>>>(kfT16, ksum);
  cvt_kvT_kernel<<<(NKV + 255) / 256, 256, 0, stream>>>(kvf, kvT16);

  // 5) num/den + normalize -> attn (f16, [B,S,D])
  num_kernel<<<fblocks, 128, 0, stream>>>(qf16, kvT16, ksum, attn16);

  // 6) output projection with bias -> f32 d_out
  gemm_f16_kernel<0><<<ggrid, 256, 0, stream>>>(attn16, WoT, bo, d_out, B_ * S_, D_, D_);

  (void)in_sizes; (void)n_in; (void)out_size; (void)ws_size;
}
